// PyramidAttention_7851200217274
// MI455X (gfx1250) — matrix operand order repaired and checked by reading, not compiled, not run on hardware
//
#include <hip/hip_runtime.h>

// ---------------------------------------------------------------------------
// PyramidAttention (spatial-reduction attention) for MI455X / gfx1250.
// B=4, C=512, H=W=64, heads=8, hd=64, SR=2.  All matmuls on bf16 WMMA
// (v_wmma_f32_16x16x32_bf16, f32 accumulate), flash-style online softmax.
// K/V chunks in the attention kernel are moved by the Tensor Data Mover
// (tensor_load_to_lds, double-buffered, s_wait_tensorcnt) when available.
// ---------------------------------------------------------------------------

typedef __attribute__((ext_vector_type(16))) __bf16 bf16x16;
typedef __attribute__((ext_vector_type(8)))  __bf16 bf16x8;
typedef __attribute__((ext_vector_type(4)))  __bf16 bf16x4;
typedef __attribute__((ext_vector_type(8)))  float  f32x8;
typedef unsigned int u32x4 __attribute__((ext_vector_type(4)));
typedef int          i32x8 __attribute__((ext_vector_type(8)));
typedef int          i32x4 __attribute__((ext_vector_type(4)));

#if defined(__has_builtin)
#if __has_builtin(__builtin_amdgcn_tensor_load_to_lds) && \
    __has_builtin(__builtin_amdgcn_s_wait_tensorcnt)
#define HAVE_TDM 1
#endif
#endif
#ifndef HAVE_TDM
#define HAVE_TDM 0
#endif

#define DEV static __device__ __forceinline__

DEV bf16x16 cat8(bf16x8 lo, bf16x8 hi) {
  return __builtin_shufflevector(lo, hi, 0,1,2,3,4,5,6,7,8,9,10,11,12,13,14,15);
}

// A-fragment (16x32 bf16, MxK): lane l -> m=l%16, half=l/16.
// elements 0..15 = K[16h .. 16h+15] (contiguous)
DEV bf16x16 load_afrag(const __bf16* row, int half) {
  bf16x8 lo = *(const bf16x8*)(row + 16 * half);
  bf16x8 hi = *(const bf16x8*)(row + 16 * half + 8);
  return cat8(lo, hi);
}
// B-fragment (32x16 bf16, KxN): lane l -> n=l%16, half=l/16.
// elements 0..15 = K[16h .. 16h+15] (contiguous)
DEV bf16x16 load_bfrag(const __bf16* row, int half) {
  const __bf16* p = row + 16 * half;
  return cat8(*(const bf16x8*)p, *(const bf16x8*)(p + 8));
}
DEV f32x8 wmma_bf16(bf16x16 a, bf16x16 b, f32x8 c) {
  // (neg_a, A, neg_b, B, c_mod, C, reuse_a, reuse_b)
  return __builtin_amdgcn_wmma_f32_16x16x32_bf16(false, a, false, b, (short)0, c, false, false);
}

DEV float hmax16(float v) {   // reduce over 16-lane half (D layout n-dimension)
  v = fmaxf(v, __shfl_xor(v, 1, 32));
  v = fmaxf(v, __shfl_xor(v, 2, 32));
  v = fmaxf(v, __shfl_xor(v, 4, 32));
  v = fmaxf(v, __shfl_xor(v, 8, 32));
  return v;
}
DEV float hsum16(float v) {
  v += __shfl_xor(v, 1, 32);
  v += __shfl_xor(v, 2, 32);
  v += __shfl_xor(v, 4, 32);
  v += __shfl_xor(v, 8, 32);
  return v;
}
DEV float wsum32(float v) {
  v += __shfl_xor(v, 1, 32);  v += __shfl_xor(v, 2, 32);
  v += __shfl_xor(v, 4, 32);  v += __shfl_xor(v, 8, 32);
  v += __shfl_xor(v, 16, 32);
  return v;
}

#if HAVE_TDM
// Issue one 2D TDM tile load (bf16 elements) global -> LDS.
//   ldsAddr      : LDS byte offset of tile destination
//   gptr         : global byte address of tile start
//   dim0/dim1    : tensor extents  (== tile extents here, data_size units)
//   tile0/tile1  : tile extents    (dim0 = contiguous/fast dimension)
//   stride0      : elements between consecutive dim1 lines
//   padIntvCode  : pad after every 2^(code+1) DWORDs   (== one line)
// Pads 4 DWORDs (16B) per line so LDS pitch stays 16B-aligned & conflict-light.
DEV void tdm_load_2d_bf16(unsigned ldsAddr, const void* gptr,
                          int dim0, int dim1, int tile0, int tile1,
                          int stride0, int padIntvCode) {
  unsigned long long ga = (unsigned long long)(uintptr_t)gptr;
  u32x4 g0;
  g0[0] = 1u;                                   // count=1, user mode, no gather
  g0[1] = ldsAddr;                              // lds_addr
  g0[2] = (unsigned)(ga & 0xFFFFFFFFu);         // global_addr[31:0]
  g0[3] = (unsigned)((ga >> 32) & 0x01FFFFFFu)  // global_addr[56:32]
          | 0x80000000u;                        // type = 2 ("image")
  i32x8 g1;
  g1[0] = (1 << 16)                             // data_size = 1 (2 bytes)
          | (1 << 20)                           // pad_enable
          | (padIntvCode << 22)                 // pad_interval
          | (3 << 25);                          // pad_amount = 4 DWORDs (16B)
  g1[1] = (dim0 & 0xFFFF) << 16;                // tensor_dim0[15:0] @ [63:48]
  g1[2] = ((dim0 >> 16) & 0xFFFF) | ((dim1 & 0xFFFF) << 16);
  g1[3] = ((dim1 >> 16) & 0xFFFF) | ((tile0 & 0xFFFF) << 16);
  g1[4] = (tile1 & 0xFFFF);                     // tile_dim1, tile_dim2=0
  g1[5] = stride0;                              // tensor_dim0_stride[31:0]
  g1[6] = 0;
  g1[7] = 0;
  i32x4 gz4 = {0, 0, 0, 0};
  i32x8 gz8 = {0, 0, 0, 0, 0, 0, 0, 0};
  // clang-23 / therock lane: 6-arg form (g0, g1, g2, g3, <reserved>, cpol)
  __builtin_amdgcn_tensor_load_to_lds(g0, g1, gz4, gz4, gz8, 0);
}
#endif

// ---------------------------------------------------------------------------
// Generic bf16 WMMA GEMM:  C[m][n] = sum_k A[m][k] * B_b[n][k]   (both K-major)
// Block = 256 threads = 8 waves (2x4); each wave computes a 64x64 tile
// (4x4 WMMA tiles, 128 accumulator VGPRs), fragments streamed from global
// (whole working set lives in the 192MB L2).
// ---------------------------------------------------------------------------
enum { OUT_TBF16 = 0, OUT_TF32 = 1, OUT_NF32 = 2 };

template <int MODE>
__global__ __launch_bounds__(256)
void gemm_wmma(const __bf16* __restrict__ A, const __bf16* __restrict__ Bmat,
               const float* __restrict__ bias, float scale, void* __restrict__ Out,
               int M, int N, int K, long bStrideB, long bStrideO) {
  const int wave = threadIdx.x >> 5;
  const int lane = threadIdx.x & 31;
  const int half = lane >> 4;
  const int l16  = lane & 15;
  const int m0 = blockIdx.x * 256 + (wave >> 1) * 64;
  const int n0 = blockIdx.y * 128 + (wave & 1) * 64;
  const __bf16* Bb = Bmat + (size_t)blockIdx.z * bStrideB;

  const __bf16* arow[4];
  const __bf16* brow[4];
#pragma unroll
  for (int mt = 0; mt < 4; ++mt) arow[mt] = A  + (size_t)(m0 + mt * 16 + l16) * K;
#pragma unroll
  for (int nt = 0; nt < 4; ++nt) brow[nt] = Bb + (size_t)(n0 + nt * 16 + l16) * K;

  f32x8 acc[4][4] = {};

  for (int k0 = 0; k0 < K; k0 += 32) {
    bf16x16 af[4], bfv[4];
#pragma unroll
    for (int mt = 0; mt < 4; ++mt) af[mt]  = load_afrag(arow[mt] + k0, half);
#pragma unroll
    for (int nt = 0; nt < 4; ++nt) bfv[nt] = load_bfrag(brow[nt] + k0, half);
#pragma unroll
    for (int mt = 0; mt < 4; ++mt)
#pragma unroll
      for (int nt = 0; nt < 4; ++nt)
        acc[mt][nt] = wmma_bf16(af[mt], bfv[nt], acc[mt][nt]);
  }

#pragma unroll
  for (int mt = 0; mt < 4; ++mt) {
    const int mbase = m0 + mt * 16 + 8 * half;   // D: VGPR r -> m = mbase + r
    float4 b0 = *(const float4*)(bias + mbase);
    float4 b1 = *(const float4*)(bias + mbase + 4);
    float bb[8] = {b0.x, b0.y, b0.z, b0.w, b1.x, b1.y, b1.z, b1.w};
    if (MODE == OUT_TBF16) {
      __bf16* O = (__bf16*)Out + (size_t)blockIdx.z * bStrideO;
#pragma unroll
      for (int nt = 0; nt < 4; ++nt) {
        const int n = n0 + nt * 16 + l16;
        bf16x8 v;
#pragma unroll
        for (int r = 0; r < 8; ++r) v[r] = (__bf16)((acc[mt][nt][r] + bb[r]) * scale);
        *(bf16x8*)(O + (size_t)n * M + mbase) = v;
      }
    } else if (MODE == OUT_TF32) {
      float* O = (float*)Out + (size_t)blockIdx.z * bStrideO;
#pragma unroll
      for (int nt = 0; nt < 4; ++nt) {
        const int n = n0 + nt * 16 + l16;
        float4 lo4 = make_float4(acc[mt][nt][0] + bb[0], acc[mt][nt][1] + bb[1],
                                 acc[mt][nt][2] + bb[2], acc[mt][nt][3] + bb[3]);
        float4 hi4 = make_float4(acc[mt][nt][4] + bb[4], acc[mt][nt][5] + bb[5],
                                 acc[mt][nt][6] + bb[6], acc[mt][nt][7] + bb[7]);
        *(float4*)(O + (size_t)n * M + mbase)     = lo4;
        *(float4*)(O + (size_t)n * M + mbase + 4) = hi4;
      }
    } else {  // OUT_NF32
      float* O = (float*)Out + (size_t)blockIdx.z * bStrideO;
#pragma unroll
      for (int r = 0; r < 8; ++r) {
        const int m = mbase + r;
#pragma unroll
        for (int nt = 0; nt < 4; ++nt)
          O[(size_t)m * N + n0 + nt * 16 + l16] = acc[mt][nt][r] + bb[r];
      }
    }
  }
}

// ---------------------------------------------------------------------------
// Flash attention: one wave per 64-query block per (batch,head).
// qT [b][4096][512] bf16 (pre-scaled by hd^-0.5), kvT [b][1024][1024] bf16
// (ch 0..511 = K), vT [b][512][1024] bf16 (channel-major), out attO
// [b][4096][512] bf16.  Streams 32 keys/chunk: 16 WMMA QK^T + 16 WMMA PV.
// TDM path: K-tile (32x64) and V-tile (64x32) are DMA'd to LDS with padded
// pitch (144B / 80B), double-buffered one chunk ahead (s_wait_tensorcnt 2).
// ---------------------------------------------------------------------------
__global__ __launch_bounds__(32)
void flash_attn(const __bf16* __restrict__ qT, const __bf16* __restrict__ kvT,
                const __bf16* __restrict__ vT, __bf16* __restrict__ attO) {
  const int lane = threadIdx.x;
  const int half = lane >> 4;
  const int l16  = lane & 15;
  const int q0   = blockIdx.x * 64;
  const int bh   = blockIdx.y;
  const int b    = bh >> 3;
  const int h    = bh & 7;

  __shared__ __align__(16) __bf16 Plds[64][40];   // P tile, 80B pitch

#if HAVE_TDM
  // K tile: 32 rows x 64 ch, line 128B + 16B pad -> pitch 144B (72 bf16)
  // V tile: 64 rows x 32 keys, line 64B + 16B pad -> pitch 80B (40 bf16)
  __shared__ __align__(16) __bf16 Kbuf[2][32 * 72];
  __shared__ __align__(16) __bf16 Vbuf[2][64 * 40];
  const __bf16* kbase = kvT + ((size_t)b * 1024) * 1024 + h * 64;
  const __bf16* vbase = vT + ((size_t)b * 512 + h * 64) * 1024;
#endif

  // Persistent Q fragments: 4 m-tiles x 2 k-steps (hd = 64 = 2*32)
  bf16x16 qf[4][2];
#pragma unroll
  for (int mt = 0; mt < 4; ++mt) {
    const __bf16* qrow = qT + ((size_t)b * 4096 + q0 + mt * 16 + l16) * 512 + h * 64;
#pragma unroll
    for (int ks = 0; ks < 2; ++ks) qf[mt][ks] = load_afrag(qrow + ks * 32, half);
  }

  float Mx[4][8], L[4][8];
  f32x8 Oacc[4][4] = {};
#pragma unroll
  for (int mt = 0; mt < 4; ++mt)
#pragma unroll
    for (int r = 0; r < 8; ++r) { Mx[mt][r] = -3.0e38f; L[mt][r] = 0.0f; }

#if HAVE_TDM
  // Prefetch chunk 0 into buffer 0.
  tdm_load_2d_bf16((unsigned)(uintptr_t)(void*)&Kbuf[0][0], kbase,
                   64, 32, 64, 32, 1024, 4);
  tdm_load_2d_bf16((unsigned)(uintptr_t)(void*)&Vbuf[0][0], vbase,
                   32, 64, 32, 64, 1024, 3);
#endif

  for (int chunk = 0; chunk < 32; ++chunk) {
    const int n0c = chunk * 32;
    const int cur = chunk & 1;

    // --- S = Q K^T over this 32-key chunk ---
    bf16x16 kf[2][2];
#if HAVE_TDM
    if (chunk < 31) {   // prefetch next chunk into the other buffer
      tdm_load_2d_bf16((unsigned)(uintptr_t)(void*)&Kbuf[cur ^ 1][0],
                       kbase + (size_t)(n0c + 32) * 1024, 64, 32, 64, 32, 1024, 4);
      tdm_load_2d_bf16((unsigned)(uintptr_t)(void*)&Vbuf[cur ^ 1][0],
                       vbase + (n0c + 32), 32, 64, 32, 64, 1024, 3);
      __builtin_amdgcn_s_wait_tensorcnt(2);   // current chunk's 2 ops done
    } else {
      __builtin_amdgcn_s_wait_tensorcnt(0);
    }
    const __bf16* Kb = &Kbuf[cur][0];
    const __bf16* Vb = &Vbuf[cur][0];
#pragma unroll
    for (int nt = 0; nt < 2; ++nt) {
      const __bf16* krow = Kb + (nt * 16 + l16) * 72;   // 144B pitch
#pragma unroll
      for (int ks = 0; ks < 2; ++ks) kf[nt][ks] = load_bfrag(krow + ks * 32, half);
    }
#else
#pragma unroll
    for (int nt = 0; nt < 2; ++nt) {
      const __bf16* krow = kvT + ((size_t)b * 1024 + n0c + nt * 16 + l16) * 1024 + h * 64;
#pragma unroll
      for (int ks = 0; ks < 2; ++ks) kf[nt][ks] = load_bfrag(krow + ks * 32, half);
    }
#endif
    f32x8 S[4][2] = {};
#pragma unroll
    for (int ks = 0; ks < 2; ++ks)
#pragma unroll
      for (int mt = 0; mt < 4; ++mt)
#pragma unroll
        for (int nt = 0; nt < 2; ++nt)
          S[mt][nt] = wmma_bf16(qf[mt][ks], kf[nt][ks], S[mt][nt]);

    // --- online softmax (row m = mt*16 + r + 8*half, cols across 16 lanes) ---
#pragma unroll
    for (int mt = 0; mt < 4; ++mt) {
#pragma unroll
      for (int r = 0; r < 8; ++r) {
        float cm = hmax16(fmaxf(S[mt][0][r], S[mt][1][r]));
        float mn = fmaxf(Mx[mt][r], cm);
        float al = __expf(Mx[mt][r] - mn);
        Mx[mt][r] = mn;
        float p0 = __expf(S[mt][0][r] - mn);
        float p1 = __expf(S[mt][1][r] - mn);
        S[mt][0][r] = p0;
        S[mt][1][r] = p1;
        L[mt][r] = L[mt][r] * al + hsum16(p0 + p1);
#pragma unroll
        for (int ct = 0; ct < 4; ++ct) Oacc[mt][ct][r] *= al;
        const int row = mt * 16 + r + 8 * half;
        Plds[row][l16]      = (__bf16)p0;
        Plds[row][16 + l16] = (__bf16)p1;
      }
    }

    // --- O += P V^T  (A = P from LDS transpose, B = V rows, contiguous) ---
    bf16x16 pf[4];
#pragma unroll
    for (int mt = 0; mt < 4; ++mt) pf[mt] = load_afrag(&Plds[mt * 16 + l16][0], half);
    bf16x16 vf[4];
#pragma unroll
    for (int ct = 0; ct < 4; ++ct) {
#if HAVE_TDM
      vf[ct] = load_bfrag(Vb + (ct * 16 + l16) * 40, half);   // 80B pitch
#else
      const __bf16* vrow = vT + ((size_t)b * 512 + h * 64 + ct * 16 + l16) * 1024;
      vf[ct] = load_bfrag(vrow + n0c, half);
#endif
    }
#pragma unroll
    for (int mt = 0; mt < 4; ++mt)
#pragma unroll
      for (int ct = 0; ct < 4; ++ct)
        Oacc[mt][ct] = wmma_bf16(pf[mt], vf[ct], Oacc[mt][ct]);
  }

  // --- normalize & store: attO[b][p][h*64+c] ---
#pragma unroll
  for (int mt = 0; mt < 4; ++mt)
#pragma unroll
    for (int r = 0; r < 8; ++r) {
      const float inv = 1.0f / L[mt][r];
      const int p = q0 + mt * 16 + r + 8 * half;
#pragma unroll
      for (int ct = 0; ct < 4; ++ct) {
        const int c = h * 64 + ct * 16 + l16;
        attO[((size_t)b * 4096 + p) * 512 + c] = (__bf16)(Oacc[mt][ct][r] * inv);
      }
    }
}

// ---------------------------------------------------------------------------
// Helper kernels
// ---------------------------------------------------------------------------
__global__ void f2bf(const float* __restrict__ in, __bf16* __restrict__ out, int n) {
  int i = blockIdx.x * 256 + threadIdx.x;
  if (i < n) out[i] = (__bf16)in[i];
}

// x [b][512][4096] f32  ->  xT [b][4096][512] bf16   (32x32 LDS tile transpose)
__global__ __launch_bounds__(256)
void transpose_x(const float* __restrict__ x, __bf16* __restrict__ xT) {
  __shared__ float t[32][33];
  const int b  = blockIdx.z;
  const int p0 = blockIdx.x * 32;
  const int c0 = blockIdx.y * 32;
  const int tx = threadIdx.x & 31;
  const int ty = threadIdx.x >> 5;       // 0..7
#pragma unroll
  for (int i = 0; i < 4; ++i) {
    const int c = c0 + ty + i * 8;
    t[ty + i * 8][tx] = x[((size_t)b * 512 + c) * 4096 + p0 + tx];
  }
  __syncthreads();
#pragma unroll
  for (int i = 0; i < 4; ++i) {
    const int p = p0 + ty + i * 8;
    xT[((size_t)b * 4096 + p) * 512 + c0 + tx] = (__bf16)t[tx][ty + i * 8];
  }
}

// im2col for 2x2/stride2 conv:  col[b][p][c*4+u*2+v] = x[b][c][2i+u][2j+v]
__global__ void im2col_ds(const float* __restrict__ x, __bf16* __restrict__ col) {
  const int g = blockIdx.x * 256 + threadIdx.x;   // 4*512*1024 threads
  const int p = g & 1023;
  const int c = (g >> 10) & 511;
  const int b = g >> 19;
  const int i = p >> 5, j = p & 31;
  const float* xb = x + (((size_t)b * 512 + c) * 64 + 2 * i) * 64 + 2 * j;
  float2 r0 = *(const float2*)(xb);
  float2 r1 = *(const float2*)(xb + 64);
  bf16x4 o;
  o[0] = (__bf16)r0.x; o[1] = (__bf16)r0.y;
  o[2] = (__bf16)r1.x; o[3] = (__bf16)r1.y;
  *(bf16x4*)(col + ((size_t)b * 1024 + p) * 2048 + c * 4) = o;
}

// channel LayerNorm over contiguous 512-f32 rows -> bf16 rows. 1 wave/row.
__global__ __launch_bounds__(256)
void layernorm_rows(const float* __restrict__ xds, const float* __restrict__ lw,
                    const float* __restrict__ lb, __bf16* __restrict__ out) {
  const int row  = blockIdx.x * 8 + (threadIdx.x >> 5);
  const int lane = threadIdx.x & 31;
  const float* r = xds + (size_t)row * 512;
  float v[16], s = 0.0f, s2 = 0.0f;
#pragma unroll
  for (int i = 0; i < 16; ++i) {
    v[i] = r[lane + i * 32];
    s += v[i];
    s2 += v[i] * v[i];
  }
  s  = wsum32(s);
  s2 = wsum32(s2);
  const float mean = s * (1.0f / 512.0f);
  const float var  = s2 * (1.0f / 512.0f) - mean * mean;
  const float rs   = rsqrtf(var + 1e-5f);
  __bf16* o = out + (size_t)row * 512;
#pragma unroll
  for (int i = 0; i < 16; ++i) {
    const int idx = lane + i * 32;
    o[idx] = (__bf16)((v[i] - mean) * rs * lw[idx] + lb[idx]);
  }
}

// vT[b][c][n] = kvT[b][n][512+c]   (bf16 32x32 tile transpose)
__global__ __launch_bounds__(256)
void transpose_v(const __bf16* __restrict__ kvT, __bf16* __restrict__ vT) {
  __shared__ __bf16 t[32][34];
  const int b  = blockIdx.z;
  const int n0 = blockIdx.x * 32;
  const int c0 = blockIdx.y * 32;
  const int tx = threadIdx.x & 31;
  const int ty = threadIdx.x >> 5;
#pragma unroll
  for (int i = 0; i < 4; ++i) {
    const int n = n0 + ty + i * 8;
    t[ty + i * 8][tx] = kvT[((size_t)b * 1024 + n) * 1024 + 512 + c0 + tx];
  }
  __syncthreads();
#pragma unroll
  for (int i = 0; i < 4; ++i) {
    const int c = c0 + ty + i * 8;
    vT[((size_t)b * 512 + c) * 1024 + n0 + tx] = t[tx][ty + i * 8];
  }
}

// ---------------------------------------------------------------------------
extern "C" void kernel_launch(void* const* d_in, const int* in_sizes, int n_in,
                              void* d_out, int out_size, void* d_ws, size_t ws_size,
                              hipStream_t stream) {
  (void)in_sizes; (void)n_in; (void)out_size; (void)ws_size;
  const float* x      = (const float*)d_in[0];
  const float* q_w    = (const float*)d_in[1];
  const float* q_b    = (const float*)d_in[2];
  const float* kv_w   = (const float*)d_in[3];
  const float* kv_b   = (const float*)d_in[4];
  const float* ds_w   = (const float*)d_in[5];
  const float* ds_b   = (const float*)d_in[6];
  const float* ln_w   = (const float*)d_in[7];
  const float* ln_b   = (const float*)d_in[8];
  const float* proj_w = (const float*)d_in[9];
  const float* proj_b = (const float*)d_in[10];

  char* w = (char*)d_ws;
  __bf16* wq    = (__bf16*)w; w += (size_t)512 * 512 * 2;
  __bf16* wkv   = (__bf16*)w; w += (size_t)1024 * 512 * 2;
  __bf16* wproj = (__bf16*)w; w += (size_t)512 * 512 * 2;
  __bf16* wds   = (__bf16*)w; w += (size_t)512 * 2048 * 2;
  __bf16* xT    = (__bf16*)w; w += (size_t)4 * 4096 * 512 * 2;
  __bf16* col   = (__bf16*)w; w += (size_t)4 * 1024 * 2048 * 2;
  float*  xds   = (float*)w;  w += (size_t)4 * 1024 * 512 * 4;
  __bf16* xdsln = (__bf16*)w; w += (size_t)4 * 1024 * 512 * 2;
  __bf16* qT    = (__bf16*)w; w += (size_t)4 * 4096 * 512 * 2;
  __bf16* kvT   = (__bf16*)w; w += (size_t)4 * 1024 * 1024 * 2;
  __bf16* vT    = (__bf16*)w; w += (size_t)4 * 512 * 1024 * 2;
  __bf16* attO  = (__bf16*)w; w += (size_t)4 * 4096 * 512 * 2;

  // 1. weights -> bf16
  f2bf<<<(512 * 512 + 255) / 256, 256, 0, stream>>>(q_w, wq, 512 * 512);
  f2bf<<<(1024 * 512 + 255) / 256, 256, 0, stream>>>(kv_w, wkv, 1024 * 512);
  f2bf<<<(512 * 512 + 255) / 256, 256, 0, stream>>>(proj_w, wproj, 512 * 512);
  f2bf<<<(512 * 2048 + 255) / 256, 256, 0, stream>>>(ds_w, wds, 512 * 2048);

  // 2. activation layouts
  transpose_x<<<dim3(128, 16, 4), 256, 0, stream>>>(x, xT);
  im2col_ds<<<(4 * 512 * 1024) / 256, 256, 0, stream>>>(x, col);

  // 3. q = q_w @ x  (scale hd^-0.5 folded in)  -> qT [b][4096][512] bf16
  gemm_wmma<OUT_TBF16><<<dim3(2, 32, 4), 256, 0, stream>>>(
      wq, xT, q_b, 0.125f, qT, 512, 4096, 512, 4096L * 512, 4096L * 512);

  // 4. strided conv as GEMM -> xds [b][1024][512] f32 (pre-LN, bias ds_b)
  gemm_wmma<OUT_TF32><<<dim3(2, 8, 4), 256, 0, stream>>>(
      wds, col, ds_b, 1.0f, xds, 512, 1024, 2048, 1024L * 2048, 1024L * 512);

  // 5. channel LayerNorm -> xdsln bf16
  layernorm_rows<<<512, 256, 0, stream>>>(xds, ln_w, ln_b, xdsln);

  // 6. kv = kv_w @ xdsln -> kvT [b][1024][1024] bf16 (0..511 = K, 512..1023 = V)
  gemm_wmma<OUT_TBF16><<<dim3(4, 8, 4), 256, 0, stream>>>(
      wkv, xdsln, kv_b, 1.0f, kvT, 1024, 1024, 512, 1024L * 512, 1024L * 1024);

  // 7. V -> channel-major vT [b][512][1024]
  transpose_v<<<dim3(32, 16, 4), 256, 0, stream>>>(kvT, vT);

  // 8. flash attention -> attO [b][4096][512] bf16
  flash_attn<<<dim3(64, 32), 32, 0, stream>>>(qT, kvT, vT, attO);

  // 9. out-proj -> d_out [b][512][4096] f32
  gemm_wmma<OUT_NF32><<<dim3(2, 32, 4), 256, 0, stream>>>(
      wproj, attO, proj_b, 1.0f, (float*)d_out, 512, 4096, 512, 4096L * 512, 4096L * 512);
}
